// CustomBiGRU_62027917689189
// MI455X (gfx1250) — matrix operand order repaired and checked by reading, not compiled, not run on hardware
//
#include <hip/hip_runtime.h>
#include <hip/hip_bf16.h>

// ---------------------------------------------------------------------------
// CustomBiGRU on MI455X (gfx1250): bf16 WMMA for all GEMMs.
//
//   T=512, B=64, IN=512, H=512, L=2.  3H=1536 (gate order r,z,n).
//
// Input flattening (setup_inputs dict insertion order):
//   d_in[0] = x (T,B,IN) f32
//   then for layer l in {0,1}, dir d in {fwd,bwd}:
//     W_ih (in_l,1536), b_ih(1536), W_rzh(512,1024), W_nh(512,512), b_nh(512)
//
// d_out = output (T,B,2H) f32  ++  hidden (4,B,H) f32
// ---------------------------------------------------------------------------

#define TT 512
#define BB 64
#define INW 512
#define HH 512
#define N3 1536          // 3*H
#define NT3 96           // N-tiles of 1536
#define FRAG 512         // bf16 elements per packed B fragment (1 KB)
#define OUT_ELEMS (TT * BB * 2 * HH)   // 33,554,432

typedef __bf16 bf16;
typedef __attribute__((ext_vector_type(8)))  __bf16 v8bf;
typedef __attribute__((ext_vector_type(16))) __bf16 v16bf;
typedef __attribute__((ext_vector_type(8)))  float  v8f;
typedef __attribute__((ext_vector_type(4)))  float  v4f;

// ------------------------------ utility kernels -----------------------------

__global__ __launch_bounds__(256)
void cast_f32_to_bf16(const float* __restrict__ src, bf16* __restrict__ dst, long n) {
    long i = (long)blockIdx.x * blockDim.x + threadIdx.x;
    if (i < n) dst[i] = (bf16)src[i];
}

// Pack a row-major f32 weight (K x Nsub) into bf16 WMMA B-fragments.
// Fragment grid: Ktiles x NT3 (row-major), each fragment = 512 bf16 (1 KB):
//   lane = ((ki>>3)&1)*16 + ni,  slot j = (ki & 7) + ((ki>>4)<<3)  (K pairs per dword).
__global__ __launch_bounds__(256)
void pack_wmma_b(const float* __restrict__ W, bf16* __restrict__ dst,
                 int K, int Nsub, int nt0) {
    long idx = (long)blockIdx.x * blockDim.x + threadIdx.x;
    long total = (long)K * Nsub;
    if (idx >= total) return;
    int k = (int)(idx / Nsub);
    int n = (int)(idx % Nsub);
    int kt = k >> 5, ki = k & 31;
    int nt = n >> 4, ni = n & 15;
    int lane = ((ki >> 3) & 1) * 16 + ni;
    int j = (ki & 7) + ((ki >> 4) << 3);
    long di = ((long)(kt * NT3 + (nt0 + nt)) << 9) + lane * 16 + j;
    dst[di] = (bf16)W[idx];
}

// --------------------------- WMMA fragment helpers --------------------------

__device__ __forceinline__ v16bf load_a_frag(const bf16* rowbase, int lane) {
    // 16-bit A layout: lane<16 -> K {0..7,16..23}; lane>=16 -> K {8..15,24..31}
    int g0 = (lane >= 16) ? 8 : 0;
    v8bf lo = *(const v8bf*)(rowbase + g0);
    v8bf hi = *(const v8bf*)(rowbase + g0 + 16);
    return __builtin_shufflevector(lo, hi, 0,1,2,3,4,5,6,7,8,9,10,11,12,13,14,15);
}

__device__ __forceinline__ v8f wmma_bf16(v16bf a, v16bf b, v8f c) {
    return __builtin_amdgcn_wmma_f32_16x16x32_bf16(
        /*neg_a=*/false, a, /*neg_b=*/false, b,
        /*c_mod=*/(short)0, c, /*reuse_a=*/false, /*reuse_b=*/false);
}

// ------------------------- pre-projection GEMM ------------------------------
// C(M x 1536) = A(M x K, bf16 row-major) @ Bpacked + bias(f32), stored bf16.
// Wave computes a 16x64 strip (4 N-tiles).  One-deep software pipeline:
// each WMMA consumes a fragment loaded one slot earlier; q-offsets are
// instruction immediates, the B pointer advances by a fixed 96 KB per K-step.
template <int KTILES>
__global__ __launch_bounds__(256)
void gemm_xproj(const bf16* __restrict__ A, const bf16* __restrict__ Bp,
                const float* __restrict__ bias, bf16* __restrict__ C,
                int Mtiles) {
    const int ngr = NT3 >> 2;              // 24 groups of 4 N-tiles
    int gw   = blockIdx.x * (blockDim.x >> 5) + (threadIdx.x >> 5);
    int lane = threadIdx.x & 31;
    if (gw >= Mtiles * ngr) return;        // wave-uniform
    int mt  = gw / ngr;
    int nt0 = (gw % ngr) * 4;

    const bf16* ak = A + (long)(mt * 16 + (lane & 15)) * (KTILES * 32);
    const bf16* bk = Bp + ((long)nt0 << 9) + lane * 16;

    v8f zero = {};
    v8f acc[4];
#pragma unroll
    for (int q = 0; q < 4; ++q) acc[q] = zero;

    v16bf av = load_a_frag(ak, lane);
    v16bf bc = *(const v16bf*)bk;
    for (int kt = 0; kt < KTILES; ++kt) {
        const bf16* bk1 = bk + (long)NT3 * FRAG;
        v16bf av_n = av;
        if (kt < KTILES - 1) av_n = load_a_frag(ak + 32, lane);
#pragma unroll
        for (int q = 0; q < 4; ++q) {
            v16bf bn = bc;
            if (q < 3)                bn = *(const v16bf*)(bk + (q + 1) * FRAG);
            else if (kt < KTILES - 1) bn = *(const v16bf*)bk1;
            acc[q] = wmma_bf16(av, bc, acc[q]);
            bc = bn;
        }
        av = av_n;
        ak += 32;
        bk = bk1;
    }

    // C/D layout: VGPR r -> row mt*16 + r + (lane>=16?8:0), col nt*16 + (lane&15)
    int r0 = mt * 16 + ((lane >= 16) ? 8 : 0);
#pragma unroll
    for (int q = 0; q < 4; ++q) {
        int col = (nt0 + q) * 16 + (lane & 15);
        float bv = bias ? bias[col] : 0.0f;
#pragma unroll
        for (int r = 0; r < 8; ++r)
            C[(long)(r0 + r) * N3 + col] = (bf16)(acc[q][r] + bv);
    }
}

// ------------------------------ recurrent cell ------------------------------
// grid = (B/16 batch chunks, 2 directions); 512 threads = 16 waves (4/SIMD).
// Per step: G(16x1536) = h_bf16(16x512) @ [W_rzh|W_nh].  K-outer with 6
// accumulator tiles per wave + 1-deep fragment pipeline.  The weight *offset*
// (not the pointer) is laundered through an empty asm each step: LICM cannot
// hoist the time-invariant fragment loads (no spill), while pointer
// provenance from the kernel arg keeps them global_load (LOADcnt only).
#define CELL_THREADS 512
#define TPW 6            // N-tiles per wave (96 / 16 waves)
__global__ __launch_bounds__(CELL_THREADS)
void bigru_cell(const bf16* __restrict__ xp_f, const bf16* __restrict__ xp_b,
                const bf16* __restrict__ Wp_f, const bf16* __restrict__ Wp_b,
                const float* __restrict__ bnh_f, const float* __restrict__ bnh_b,
                float* __restrict__ out_f32,   // (T,B,2H) or nullptr
                bf16* __restrict__ out_bf16,   // (T,B,2H) or nullptr
                float* __restrict__ hidden,    // (4,B,H)
                int layer) {
    __shared__ float hG[16 * N3];     //  96 KB: gate pre-activations
    __shared__ bf16  hbf[16 * HH];    //  16 KB: h in bf16 (WMMA A operand)
    __shared__ float hf [16 * HH];    //  32 KB: h in f32

    const int dir = blockIdx.y;
    const int b0  = blockIdx.x * 16;
    const bf16*  xp  = dir ? xp_b  : xp_f;
    const bf16*  Wp  = dir ? Wp_b  : Wp_f;
    const float* bnh = dir ? bnh_b : bnh_f;

    const int tid  = threadIdx.x;
    const int lane = tid & 31;
    const int wv   = tid >> 5;           // 0..15

    for (int i = tid; i < 16 * HH; i += CELL_THREADS) { hf[i] = 0.0f; hbf[i] = (bf16)0.0f; }
    __syncthreads();

    const bf16* hrow = &hbf[(lane & 15) * HH];
    const long  woff = (((long)(wv * TPW)) << 9) + lane * 16;  // wave's first fragment
    const v8f   zero = {};

    for (int s = 0; s < TT; ++s) {
        const int t = dir ? (TT - 1 - s) : s;

        // ---- GEMM: 6 N-tiles/wave, K = 512 (16 WMMA steps), pipelined ----
        v8f acc[TPW];
#pragma unroll
        for (int q = 0; q < TPW; ++q) acc[q] = zero;

        long off = woff;
        asm volatile("" : "+v"(off));  // opaque per step: forbid hoisting W loads
        const bf16* bk = Wp + off;
        v16bf av = load_a_frag(hrow, lane);
        v16bf bc = *(const v16bf*)bk;
        for (int kt = 0; kt < 16; ++kt) {
            const bf16* bk1 = bk + (long)NT3 * FRAG;
            v16bf av_n = av;
            if (kt < 15) av_n = load_a_frag(hrow + (kt + 1) * 32, lane);
#pragma unroll
            for (int q = 0; q < TPW; ++q) {
                v16bf bn = bc;
                if (q < TPW - 1)  bn = *(const v16bf*)(bk + (q + 1) * FRAG);
                else if (kt < 15) bn = *(const v16bf*)bk1;
                acc[q] = wmma_bf16(av, bc, acc[q]);
                bc = bn;
            }
            av = av_n;
            bk = bk1;
        }
        {
            int r0 = (lane >= 16) ? 8 : 0;
#pragma unroll
            for (int q = 0; q < TPW; ++q) {
                int col = (wv * TPW + q) * 16 + (lane & 15);
#pragma unroll
                for (int r = 0; r < 8; ++r)
                    hG[(r0 + r) * N3 + col] = acc[q][r];
            }
        }
        __syncthreads();

        // ---- fused GRU gates: 16 (b,j) elements/thread, 8-wide vectors ----
        {
            int b  = tid >> 5;             // 0..15
            int j0 = (tid & 31) * 16;      // 0..496
            long xrow = ((long)t * BB + (b0 + b)) * N3;
            long orow = ((long)t * BB + (b0 + b)) * (2 * HH) + dir * HH;
#pragma unroll
            for (int c = 0; c < 2; ++c) {
                int j = j0 + c * 8;
                v8bf rx8 = *(const v8bf*)&xp[xrow + j];
                v8bf zx8 = *(const v8bf*)&xp[xrow + HH + j];
                v8bf nx8 = *(const v8bf*)&xp[xrow + 2 * HH + j];
                float hn8[8];
                v8bf  ob8;
#pragma unroll
                for (int u = 0; u < 8; ++u) {
                    int jj = j + u;
                    float rh = hG[b * N3 + jj];
                    float zh = hG[b * N3 + HH + jj];
                    float nh = hG[b * N3 + 2 * HH + jj] + bnh[jj];
                    float r  = 1.0f / (1.0f + __expf(-((float)rx8[u] + rh)));
                    float z  = 1.0f / (1.0f + __expf(-((float)zx8[u] + zh)));
                    float n  = tanhf((float)nx8[u] + r * nh);
                    float hp = hf[b * HH + jj];
                    float hn = (1.0f - z) * n + z * hp;
                    hf [b * HH + jj] = hn;
                    hbf[b * HH + jj] = (bf16)hn;
                    hn8[u] = hn;
                    ob8[u] = (bf16)hn;
                }
                if (out_f32) {
                    *(v4f*)&out_f32[orow + j]     = (v4f){hn8[0], hn8[1], hn8[2], hn8[3]};
                    *(v4f*)&out_f32[orow + j + 4] = (v4f){hn8[4], hn8[5], hn8[6], hn8[7]};
                }
                if (out_bf16) *(v8bf*)&out_bf16[orow + j] = ob8;
            }
        }
        __syncthreads();
    }

    // final hidden state: cell index = layer*2 + dir
    for (int i = tid; i < 16 * HH; i += CELL_THREADS) {
        int b = i >> 9, j = i & (HH - 1);
        hidden[((long)(layer * 2 + dir) * BB + (b0 + b)) * HH + j] = hf[b * HH + j];
    }
}

// --------------------------------- launcher ---------------------------------

extern "C" void kernel_launch(void* const* d_in, const int* in_sizes, int n_in,
                              void* d_out, int out_size, void* d_ws, size_t ws_size,
                              hipStream_t stream) {
    (void)in_sizes; (void)n_in; (void)out_size; (void)ws_size;

    const float* x = (const float*)d_in[0];
    struct Cell { const float *Wih, *bih, *Wrzh, *Wnh, *bnh; };
    Cell cell[2][2];
    {
        int idx = 1;
        for (int l = 0; l < 2; ++l)
            for (int d = 0; d < 2; ++d) {
                cell[l][d].Wih  = (const float*)d_in[idx++];
                cell[l][d].bih  = (const float*)d_in[idx++];
                cell[l][d].Wrzh = (const float*)d_in[idx++];
                cell[l][d].Wnh  = (const float*)d_in[idx++];
                cell[l][d].bnh  = (const float*)d_in[idx++];
            }
    }

    const long M = (long)TT * BB;                  // 32768 GEMM rows
    char* ws = (char*)d_ws;
    size_t cur = 0;
    auto alloc = [&](size_t bytes) { char* p = ws + cur; cur += (bytes + 255) & ~(size_t)255; return p; };

    bf16* xbf  = (bf16*)alloc(M * 1024 * sizeof(bf16));   // layer input (max width 2H)
    bf16* xp_f = (bf16*)alloc(M * N3 * sizeof(bf16));     // fwd pre-projection
    bf16* xp_b = (bf16*)alloc(M * N3 * sizeof(bf16));     // bwd pre-projection
    bf16* WihP[2][2];
    bf16* WgP [2][2];
    for (int l = 0; l < 2; ++l)
        for (int d = 0; d < 2; ++d) {
            int in_l = (l == 0) ? INW : 2 * HH;
            WihP[l][d] = (bf16*)alloc((size_t)(in_l >> 5) * NT3 * FRAG * sizeof(bf16));
            WgP [l][d] = (bf16*)alloc((size_t)(HH >> 5)   * NT3 * FRAG * sizeof(bf16));
        }

    // ---- pack all weights into WMMA B-fragment layout ----
    for (int l = 0; l < 2; ++l)
        for (int d = 0; d < 2; ++d) {
            int in_l = (l == 0) ? INW : 2 * HH;
            long n1 = (long)in_l * N3;
            pack_wmma_b<<<(int)((n1 + 255) / 256), 256, 0, stream>>>(
                cell[l][d].Wih, WihP[l][d], in_l, N3, 0);
            long n2 = (long)HH * 2 * HH;           // W_rzh -> tiles [0,64)
            pack_wmma_b<<<(int)((n2 + 255) / 256), 256, 0, stream>>>(
                cell[l][d].Wrzh, WgP[l][d], HH, 2 * HH, 0);
            long n3 = (long)HH * HH;               // W_nh  -> tiles [64,96)
            pack_wmma_b<<<(int)((n3 + 255) / 256), 256, 0, stream>>>(
                cell[l][d].Wnh, WgP[l][d], HH, HH, 64);
        }

    // ---- layer-0 input to bf16 ----
    {
        long n = M * INW;
        cast_f32_to_bf16<<<(int)((n + 255) / 256), 256, 0, stream>>>(x, xbf, n);
    }

    const int Mtiles = (int)(M >> 4);              // 2048
    const int groups = Mtiles * (NT3 >> 2);
    const int blocks = (groups + 7) / 8;           // 8 waves per block

    for (int l = 0; l < 2; ++l) {
        if (l == 0) {
            gemm_xproj<16><<<blocks, 256, 0, stream>>>(xbf, WihP[l][0], cell[l][0].bih, xp_f, Mtiles);
            gemm_xproj<16><<<blocks, 256, 0, stream>>>(xbf, WihP[l][1], cell[l][1].bih, xp_b, Mtiles);
        } else {
            gemm_xproj<32><<<blocks, 256, 0, stream>>>(xbf, WihP[l][0], cell[l][0].bih, xp_f, Mtiles);
            gemm_xproj<32><<<blocks, 256, 0, stream>>>(xbf, WihP[l][1], cell[l][1].bih, xp_b, Mtiles);
        }

        float* out_f32  = (l == 1) ? (float*)d_out : nullptr;
        bf16*  out_bf16 = (l == 0) ? xbf : nullptr;   // layer-1 input, bf16
        float* hidden   = (float*)d_out + OUT_ELEMS;
        bigru_cell<<<dim3(BB / 16, 2), CELL_THREADS, 0, stream>>>(
            xp_f, xp_b, WgP[l][0], WgP[l][1], cell[l][0].bnh, cell[l][1].bnh,
            out_f32, out_bf16, hidden, l);
    }
}
